// WindowAttention_2413771621151
// MI455X (gfx1250) — matrix operand order repaired and checked by reading, not compiled, not run on hardware
//
#include <hip/hip_runtime.h>

// ---------------------------------------------------------------------------
// Fused window attention for MI455X (gfx1250, wave32, WMMA).
//   qkv = x @ Wqkv + b ; per-head softmax(q kT) v ; out = attn @ Wproj + b
// One workgroup (8 waves) per window; all intermediates live in LDS as bf16,
// matmuls via v_wmma_f32_16x16x32_bf16 with fp32 accumulation.
// Round-4 changes: dual N-tile register blocking (two independent WMMA
// accumulator streams per wave) in the QKV and proj GEMMs; unguarded
// prefetch chain across the contiguous Q|K|V|proj weight regions.
// ---------------------------------------------------------------------------

typedef __attribute__((ext_vector_type(16))) __bf16 v16bf;
typedef __attribute__((ext_vector_type(8)))  __bf16 v8bf;
typedef __attribute__((ext_vector_type(2)))  __bf16 v2bf;
typedef __attribute__((ext_vector_type(8)))  float  v8f;

#define NHEAD   6
#define HDIM    32
#define NTOK    49
#define CDIM    192
#define NPAD    64
#define VT_LD   80            // padded ld for V^T: spreads LDS banks, 32B-aligned
#define QK_SCALE 0.17677669529663687f   // 32^-0.5

static __device__ __forceinline__ __bf16 to_bf16(float f) { return (__bf16)f; }

static __device__ __forceinline__ unsigned pack_bf16(float lo, float hi) {
  v2bf t; t.x = (__bf16)lo; t.y = (__bf16)hi;
  unsigned u; __builtin_memcpy(&u, &t, 4);
  return u;
}

// A-operand (16x32, MxK) fragment from row-major [rows][ld] bf16:
// lanes 0-15 hold K[0..16) of row=lane; lanes 16-31 hold K[16..32).
static __device__ __forceinline__ v16bf load_A(const __bf16* base, int ld, int lane) {
  const int row = lane & 15;
  const int o1  = (lane >> 4) << 4;        // 0 or 16
  const __bf16* p = base + row * ld + o1;
  union { v16bf v; v8bf h[2]; } u;
  u.h[0] = *(const v8bf*)(p);
  u.h[1] = *(const v8bf*)(p + 8);
  return u.v;
}

// B-operand (32x16, KxN) fragment from K-contiguous ("B-transposed") [N][ld]:
// lane holds 16 contiguous K values of column = lane&15 (k base 0 or 16).
static __device__ __forceinline__ v16bf load_B(const __bf16* baseT, int ld, int lane) {
  const int col = lane & 15;
  const int k   = (lane >> 4) << 4;        // 0 or 16
  return *(const v16bf*)(baseT + col * ld + k);
}

// Two 16x16 output tiles, K=192 in 6 WMMA steps each, interleaved so the
// XDL pipe always has an independent WMMA to issue. A register-resident.
static __device__ __forceinline__ void gemm6x2(const v16bf* Af,
                                               const __bf16* wb0,
                                               const __bf16* wb1,
                                               int lane, v8f& a0, v8f& a1) {
  a0 = {}; a1 = {};
#pragma unroll
  for (int kk = 0; kk < 6; ++kk) {
    v16bf B0 = load_B(wb0 + kk * 32, CDIM, lane);
    v16bf B1 = load_B(wb1 + kk * 32, CDIM, lane);
    a0 = __builtin_amdgcn_wmma_f32_16x16x32_bf16(false, Af[kk], false, B0,
                                                 (short)0, a0, false, false);
    a1 = __builtin_amdgcn_wmma_f32_16x16x32_bf16(false, Af[kk], false, B1,
                                                 (short)0, a1, false, false);
  }
}

// ---------------------------------------------------------------------------
// Prep: fp32 weights -> transposed bf16 in workspace (L2-resident thereafter).
// wqkvT[n][k] = w_qkv[k][n]  (576x192), wprojT[n][k] = w_proj[k][n] (192x192)
// ---------------------------------------------------------------------------
__global__ void convert_weights(const float* __restrict__ w_qkv,
                                const float* __restrict__ w_proj,
                                unsigned* __restrict__ wqkvT2,
                                unsigned* __restrict__ wprojT2) {
  const int i   = blockIdx.x * blockDim.x + threadIdx.x;
  const int NQ2 = 3 * CDIM * CDIM / 2;     // pairs in wqkvT
  if (i < NQ2) {
    const int e = i * 2;
    const int n = e / CDIM, k = e % CDIM;
    wqkvT2[i] = pack_bf16(w_qkv[k * (3 * CDIM) + n], w_qkv[(k + 1) * (3 * CDIM) + n]);
  } else if (i < NQ2 + CDIM * CDIM / 2) {
    const int e = (i - NQ2) * 2;
    const int n = e / CDIM, k = e % CDIM;
    wprojT2[i - NQ2] = pack_bf16(w_proj[k * CDIM + n], w_proj[(k + 1) * CDIM + n]);
  }
}

// ---------------------------------------------------------------------------
// Fused attention: one block (8 wave32s) per window.
// ---------------------------------------------------------------------------
__global__ __launch_bounds__(256)
void attn_fused(const float* __restrict__ x,
                const float* __restrict__ b_qkv,
                const float* __restrict__ b_proj,
                const __bf16* __restrict__ wqkvT,
                const __bf16* __restrict__ wprojT,
                float* __restrict__ out) {
  __shared__ __align__(128) __bf16 xbuf[NPAD * CDIM];        // x (ph1/2), attn_out (ph3/4)
  __shared__ __align__(128) __bf16 qs[NHEAD * NPAD * HDIM];  // q (scaled), row-major [64][32]
  __shared__ __align__(128) __bf16 ks_[NHEAD * NPAD * HDIM]; // k, row-major [64][32]
  __shared__ __align__(128) __bf16 vTs[NHEAD * HDIM * VT_LD];// v^T, [32][VT_LD]
  __shared__ __align__(128) __bf16 Ps[8 * 16 * 64];          // softmax probs, per-wave [16][64]

  const int b    = blockIdx.x;
  const int tid  = threadIdx.x;
  const int wave = tid >> 5;
  const int lane = tid & 31;
  const int ln   = lane & 15;
  const int mt   = wave & 3;                 // this wave's M row-tile (phases 2/4)
  const int half = wave >> 2;                // 0/1: interleaved N-tiles
  const int rb0  = (lane >> 4) << 3;         // row offset of acc half (0 or 8)

  // ---- phase 1: stage x[b] (49x192 fp32) as bf16 pairs, zero-pad rows to 64 ----
  const float* xb = x + (size_t)b * NTOK * CDIM;
  {
    unsigned* xp = (unsigned*)xbuf;
    for (int i = tid; i < NPAD * CDIM / 2; i += 256) {   // 9408 even: pairs never straddle
      float a = 0.f, c = 0.f;
      if (i < NTOK * CDIM / 2) { a = xb[2 * i]; c = xb[2 * i + 1]; }
      xp[i] = pack_bf16(a, c);
    }
  }
  __syncthreads();

  // ---- phase 2: qkv = x @ WqkvT + b, split into Q / K / V region loops ----
  // Wave owns M-tile mt; 6 column-tiles per region processed as 3 pairs.
  {
    v16bf Af[6];
#pragma unroll
    for (int kk = 0; kk < 6; ++kk)
      Af[kk] = load_A(xbuf + mt * 16 * CDIM + kk * 32, CDIM, lane);
    const int rb = mt * 16 + rb0;

    // --- Q region: columns [0,192) -> qs (scaled) ---
    for (int j = half; j < 12; j += 4) {
      const __bf16* wb0 = wqkvT + j * 16 * CDIM;
      __builtin_prefetch(wb0 + 4 * 16 * CDIM + ln * CDIM, 0, 1);  // next pair / next region
      v8f a0, a1;
      gemm6x2(Af, wb0, wb0 + 2 * 16 * CDIM, lane, a0, a1);
#pragma unroll
      for (int p = 0; p < 2; ++p) {
        const int jj = j + 2 * p;
        v8f& acc = p ? a1 : a0;
        const float bias = b_qkv[jj * 16 + ln];
        const int head = jj >> 1;
        const int d    = ((jj & 1) << 4) + ln;
        __bf16* qd = qs + head * NPAD * HDIM + rb * HDIM + d;
#pragma unroll
        for (int r = 0; r < 8; ++r)
          qd[r * HDIM] = to_bf16((acc[r] + bias) * QK_SCALE);
      }
    }
    // --- K region: columns [192,384) -> ks_ ---
    for (int j = half; j < 12; j += 4) {
      const __bf16* wb0 = wqkvT + (12 + j) * 16 * CDIM;
      __builtin_prefetch(wb0 + 4 * 16 * CDIM + ln * CDIM, 0, 1);
      v8f a0, a1;
      gemm6x2(Af, wb0, wb0 + 2 * 16 * CDIM, lane, a0, a1);
#pragma unroll
      for (int p = 0; p < 2; ++p) {
        const int jj = j + 2 * p;
        v8f& acc = p ? a1 : a0;
        const float bias = b_qkv[CDIM + jj * 16 + ln];
        const int head = jj >> 1;
        const int d    = ((jj & 1) << 4) + ln;
        __bf16* kd = ks_ + head * NPAD * HDIM + rb * HDIM + d;
#pragma unroll
        for (int r = 0; r < 8; ++r)
          kd[r * HDIM] = to_bf16(acc[r] + bias);
      }
    }
    // --- V region: columns [384,576) -> d-major V^T, packed b32 stores ---
    for (int j = half; j < 12; j += 4) {
      const __bf16* wb0 = wqkvT + (24 + j) * 16 * CDIM;
      __builtin_prefetch(wb0 + 4 * 16 * CDIM + ln * CDIM, 0, 1);  // tail warms wprojT
      v8f a0, a1;
      gemm6x2(Af, wb0, wb0 + 2 * 16 * CDIM, lane, a0, a1);
#pragma unroll
      for (int p = 0; p < 2; ++p) {
        const int jj = j + 2 * p;
        v8f& acc = p ? a1 : a0;
        const float bias = b_qkv[2 * CDIM + jj * 16 + ln];
        const int head = jj >> 1;
        const int d    = ((jj & 1) << 4) + ln;
        unsigned* vrow = (unsigned*)(vTs + head * HDIM * VT_LD + d * VT_LD + rb);
#pragma unroll
        for (int r = 0; r < 8; r += 2)
          vrow[r >> 1] = pack_bf16(acc[r] + bias, acc[r + 1] + bias);
      }
    }
  }
  __syncthreads();

  // ---- phase 3: per (head, row-band): S = q kT, softmax, O = P v ----
  for (int u = wave; u < NHEAD * 4; u += 8) {
    const int head = u >> 2;
    const int mt3  = u & 3;
    const __bf16* qb = qs  + head * NPAD * HDIM + mt3 * 16 * HDIM;
    const __bf16* kb = ks_ + head * NPAD * HDIM;

    v8f s[4];
    v16bf A = load_A(qb, HDIM, lane);                 // K = D = 32, single step
#pragma unroll
    for (int nt = 0; nt < 4; ++nt) {
      v16bf B = load_B(kb + nt * 16 * HDIM, HDIM, lane);
      v8f z = {};
      s[nt] = __builtin_amdgcn_wmma_f32_16x16x32_bf16(false, A, false, B,
                                                      (short)0, z, false, false);
    }
    // mask key columns >= 49 (lane%16 = column within tile)
#pragma unroll
    for (int nt = 0; nt < 4; ++nt)
      if (nt * 16 + ln >= NTOK)
#pragma unroll
        for (int r = 0; r < 8; ++r) s[nt][r] = -1e30f;

    // softmax per row: C/D layout puts row r in lanes 0-15 and row r+8 in
    // lanes 16-31 of VGPR r -> xor-shuffles 8/4/2/1 reduce each half's row.
#pragma unroll
    for (int r = 0; r < 8; ++r) {
      float m = fmaxf(fmaxf(s[0][r], s[1][r]), fmaxf(s[2][r], s[3][r]));
      m = fmaxf(m, __shfl_xor(m, 8, 32));
      m = fmaxf(m, __shfl_xor(m, 4, 32));
      m = fmaxf(m, __shfl_xor(m, 2, 32));
      m = fmaxf(m, __shfl_xor(m, 1, 32));
      float e0 = __expf(s[0][r] - m), e1 = __expf(s[1][r] - m);
      float e2 = __expf(s[2][r] - m), e3 = __expf(s[3][r] - m);
      float sum = e0 + e1 + e2 + e3;
      sum += __shfl_xor(sum, 8, 32);
      sum += __shfl_xor(sum, 4, 32);
      sum += __shfl_xor(sum, 2, 32);
      sum += __shfl_xor(sum, 1, 32);
      const float inv = __frcp_rn(sum);
      s[0][r] = e0 * inv; s[1][r] = e1 * inv; s[2][r] = e2 * inv; s[3][r] = e3 * inv;
    }

    // stash P (bf16) per-wave, then reload as A-operand (no transpose needed)
    __bf16* Pw = Ps + wave * 16 * 64;
#pragma unroll
    for (int nt = 0; nt < 4; ++nt) {
      const int colc = nt * 16 + ln;
#pragma unroll
      for (int r = 0; r < 8; ++r)
        Pw[(r + rb0) * 64 + colc] = to_bf16(s[nt][r]);
    }

    // O[16x32] = P[16x64] @ V[64x32]  (B from V^T: K-contiguous per d column)
    v8f o0 = {}, o1 = {};
#pragma unroll
    for (int kk = 0; kk < 2; ++kk) {
      v16bf Ap = load_A(Pw + kk * 32, 64, lane);
      v16bf B0 = load_B(vTs + head * HDIM * VT_LD + kk * 32, VT_LD, lane);
      v16bf B1 = load_B(vTs + head * HDIM * VT_LD + 16 * VT_LD + kk * 32, VT_LD, lane);
      o0 = __builtin_amdgcn_wmma_f32_16x16x32_bf16(false, Ap, false, B0,
                                                   (short)0, o0, false, false);
      o1 = __builtin_amdgcn_wmma_f32_16x16x32_bf16(false, Ap, false, B1,
                                                   (short)0, o1, false, false);
    }
#pragma unroll
    for (int p = 0; p < 2; ++p) {
      v8f& o = p ? o1 : o0;
      const int colg = head * HDIM + p * 16 + ln;
#pragma unroll
      for (int r = 0; r < 8; ++r)
        xbuf[(mt3 * 16 + rb0 + r) * CDIM + colg] = to_bf16(o[r]);  // attn_out reuses xbuf
    }
  }
  __syncthreads();

  // ---- phase 4: out = attn_out @ WprojT + b, fp32 store (rows < 49) ----
  float* ob = out + (size_t)b * NTOK * CDIM;
  {
    v16bf Af[6];
#pragma unroll
    for (int kk = 0; kk < 6; ++kk)
      Af[kk] = load_A(xbuf + mt * 16 * CDIM + kk * 32, CDIM, lane);
    const int rb  = mt * 16 + rb0;
    const int mts = __builtin_amdgcn_readfirstlane(mt);   // scalar: tail handling

    for (int nt = half; nt < 12; nt += 4) {
      const __bf16* wb0 = wprojT + nt * 16 * CDIM;
      if (nt + 4 < 12)
        __builtin_prefetch(wb0 + 4 * 16 * CDIM + ln * CDIM, 0, 1);
      v8f a0, a1;
      gemm6x2(Af, wb0, wb0 + 2 * 16 * CDIM, lane, a0, a1);
#pragma unroll
      for (int p = 0; p < 2; ++p) {
        const int nn = nt + 2 * p;
        v8f& acc = p ? a1 : a0;
        const int col  = nn * 16 + ln;
        const float bias = b_proj[col];
        float* od = ob + rb * CDIM + col;
        if (mts < 3) {            // rows 0..47: all valid, unguarded stores
#pragma unroll
          for (int r = 0; r < 8; ++r)
            od[r * CDIM] = acc[r] + bias;
        } else {                  // rows 48..63: only row 48 (low half, r==0) valid
          if (lane < 16) od[0] = acc[0] + bias;
        }
      }
    }
  }
}

extern "C" void kernel_launch(void* const* d_in, const int* in_sizes, int n_in,
                              void* d_out, int out_size, void* d_ws, size_t ws_size,
                              hipStream_t stream) {
  const float* x      = (const float*)d_in[0];
  const float* w_qkv  = (const float*)d_in[1];
  const float* b_qkv  = (const float*)d_in[2];
  const float* w_proj = (const float*)d_in[3];
  const float* b_proj = (const float*)d_in[4];
  float* out = (float*)d_out;

  __bf16* wqkvT  = (__bf16*)d_ws;                 // 576*192 bf16
  __bf16* wprojT = wqkvT + 3 * CDIM * CDIM;       // 192*192 bf16 (32B-aligned offset)

  const int pairs = (3 * CDIM * CDIM + CDIM * CDIM) / 2;
  convert_weights<<<(pairs + 255) / 256, 256, 0, stream>>>(
      w_qkv, w_proj, (unsigned*)wqkvT, (unsigned*)wprojT);
  attn_fused<<<4096, 256, 0, stream>>>(x, b_qkv, b_proj, wqkvT, wprojT, out);
}
